// SAttention_fast_46634754900240
// MI455X (gfx1250) — compile-verified, matrix operand order checked by reading, not run on hardware
//
#include <hip/hip_runtime.h>
#include <hip/hip_bf16.h>
#include <stdint.h>

#define NPED 1024
#define SDIM 32

typedef __attribute__((ext_vector_type(16))) __bf16 v16bf;
typedef __attribute__((ext_vector_type(8)))  float  v8f;
typedef __attribute__((ext_vector_type(4)))  float  f32x4;

struct Frag { union { v16bf v; uint32_t u[8]; }; };

__device__ __forceinline__ uint16_t f2bf(float x) {
    uint32_t u = __float_as_uint(x);
    uint32_t r = ((u >> 16) & 1u) + 0x7FFFu;   // round-to-nearest-even
    return (uint16_t)((u + r) >> 16);
}
__device__ __forceinline__ float bf2f(uint16_t h) {
    return __uint_as_float(((uint32_t)h) << 16);
}
// fast sigmoid/tanh: v_exp_f32 + v_rcp_f32, no IEEE division chains
__device__ __forceinline__ float sigf(float x) {
    return __builtin_amdgcn_rcpf(1.0f + __expf(-x));
}
__device__ __forceinline__ float tanhfast(float x) {
    return 2.0f * __builtin_amdgcn_rcpf(1.0f + __expf(-2.0f * x)) - 1.0f;
}

// ---------------- prep: fold projection chains, convert weights to bf16 ----------------
__global__ void sattn_prep(const float* __restrict__ in_w, const float* __restrict__ wq_w,
                           const float* __restrict__ wk_w, const float* __restrict__ wv_w,
                           const float* __restrict__ w_ih, const float* __restrict__ w_hh,
                           float* __restrict__ wsCq, uint16_t* __restrict__ wsCk,
                           uint16_t* __restrict__ wsCv, uint16_t* __restrict__ wsWt) {
    int tid = threadIdx.x;
    // Cq/Ck/Cv[n][s] = sum_t W{q,k,v}[n][t] * w{q,k,v}_w[t][s]
    for (int idx = tid; idx < 1024; idx += 256) {
        int n = idx >> 5, s = idx & 31;
        float cq = 0.f, ck = 0.f, cv = 0.f;
        for (int t = 0; t < 32; ++t) {
            cq += in_w[n * 32 + t]        * wq_w[t * 32 + s];
            ck += in_w[(32 + n) * 32 + t] * wk_w[t * 32 + s];
            cv += in_w[(64 + n) * 32 + t] * wv_w[t * 32 + s];
        }
        wsCq[idx] = cq;
        wsCk[idx] = f2bf(ck);
        wsCv[idx] = f2bf(cv);
    }
    // combined gate weights, stored transposed: Wt[n][k] = (k<32 ? w_ih[n][k] : w_hh[n][k-32])
    for (int idx = tid; idx < 8192; idx += 256) {
        int n = idx >> 6, k = idx & 63;
        wsWt[idx] = f2bf(k < 32 ? w_ih[n * 32 + k] : w_hh[n * 32 + (k - 32)]);
    }
}

// ---------------- hs = hidden_state @ h2s_w.T + h2s_b ----------------
__global__ void sattn_hs(const float* __restrict__ hidden_state, const float* __restrict__ h2s_w,
                         const float* __restrict__ h2s_b, float* __restrict__ hsg) {
    int gid = blockIdx.x * 256 + threadIdx.x;      // 32768 = 1024*32
    int b = gid >> 5, s = gid & 31;
    float acc = h2s_b[s];
    for (int h = 0; h < 256; ++h) acc += hidden_state[b * 256 + h] * h2s_w[s * 256 + h];
    hsg[gid] = acc;
}

// ---------------- q0 = scale * (hs @ Cq.T + bq) ----------------
__global__ void sattn_q0(const float* __restrict__ hsg, const float* __restrict__ wsCq,
                         const float* __restrict__ in_b, float* __restrict__ q0g) {
    int gid = blockIdx.x * 256 + threadIdx.x;
    int b = gid >> 5, n = gid & 31;
    float acc = in_b[n];
    for (int s = 0; s < 32; ++s) acc += hsg[b * 32 + s] * wsCq[n * 32 + s];
    q0g[gid] = 0.17677669529663687f * acc;         // 1/sqrt(32)
}

// ---------------- fused main kernel: one block per query b ----------------
__global__ __launch_bounds__(256) void sattn_main(
    const float* __restrict__ obs2,  const float* __restrict__ hc_h, const float* __restrict__ hc_c,
    const float* __restrict__ emb_w, const float* __restrict__ emb_b,
    const float* __restrict__ b_ih,  const float* __restrict__ b_hh, const float* __restrict__ in_b,
    const float* __restrict__ attn_ow, const float* __restrict__ attn_ob,
    const float* __restrict__ proj_w,  const float* __restrict__ proj_b,
    const float* __restrict__ hsg,   const float* __restrict__ q0g,
    const uint16_t* __restrict__ wsCk, const uint16_t* __restrict__ wsCv,
    const uint16_t* __restrict__ wsWt, float* __restrict__ out)
{
    const int b    = blockIdx.x;
    const int tid  = threadIdx.x;
    const int lane = tid & 31;
    const int w    = tid >> 5;
    const int half = lane >> 4;
    const int l15  = lane & 15;

    __shared__ __align__(16) uint16_t sWt[128][68];        // [n][k], bf16 combined gate weights
    __shared__ __align__(16) uint16_t sCk[32][36];         // [n][s]
    __shared__ __align__(16) uint16_t sCv[32][36];
    __shared__ float sEw0[32], sEw1[32], sEb[32];
    __shared__ float sG[32];                               // g[s] = sum_n q0[n]*Ck[n][s]
    __shared__ __align__(16) uint16_t sRelE[8][16][36];    // per-wave A tiles
    __shared__ __align__(16) uint16_t sH[8][16][36];
    __shared__ __align__(16) uint16_t sH2[8][16][36];
    __shared__ float sC[8][16][33];
    __shared__ float wMs[8], wSs[8], wCtx[8][32];

    for (int i = tid; i < 128 * 64; i += 256) sWt[i >> 6][i & 63] = wsWt[i];
    for (int i = tid; i < 32 * 32; i += 256) { sCk[i >> 5][i & 31] = wsCk[i]; sCv[i >> 5][i & 31] = wsCv[i]; }
    if (tid < 32) { sEw0[tid] = emb_w[tid * 2]; sEw1[tid] = emb_w[tid * 2 + 1]; sEb[tid] = emb_b[tid]; }
    __syncthreads();

    // ---- per-b folded query vector: scores[l] = h2[l,:].g + c0b  (kp never materialized)
    if (tid < 32) {
        float gg = 0.f;
        for (int n = 0; n < 32; ++n) gg += q0g[b * 32 + n] * bf2f(sCk[n][tid]);
        sG[tid] = gg;
    }
    float c0b = 0.f;
    for (int n = 0; n < 32; ++n) c0b += q0g[b * 32 + n] * in_b[32 + n];
    __syncthreads();

    // ---- register-resident B fragments (16-bit B layout: n = lane&15, K = (e>>3)*16 + (lane>>4)*8 + (e&7))
    const int kb = half * 8;
    Frag bW[2][8];
#pragma unroll
    for (int kt = 0; kt < 2; ++kt)
#pragma unroll
        for (int nt = 0; nt < 8; ++nt) {
            const uint16_t* rowp = &sWt[nt * 16 + l15][kt * 32];
#pragma unroll
            for (int p = 0; p < 8; ++p) {
                int K = ((p >> 2) * 16) + kb + 2 * (p & 3);
                bW[kt][nt].u[p] = *(const uint32_t*)&rowp[K];
            }
        }
    Frag bV[2];
#pragma unroll
    for (int nt = 0; nt < 2; ++nt) {
        const uint16_t* rv = &sCv[nt * 16 + l15][0];
#pragma unroll
        for (int p = 0; p < 8; ++p) {
            int K = ((p >> 2) * 16) + kb + 2 * (p & 3);
            bV[nt].u[p] = *(const uint32_t*)&rv[K];
        }
    }
    // score B fragment: only column n==0 nonzero, holds g[s] (bf16)
    Frag bG;
#pragma unroll
    for (int p = 0; p < 8; ++p) {
        int K = ((p >> 2) * 16) + kb + 2 * (p & 3);
        float g0 = (l15 == 0) ? sG[K] : 0.f;
        float g1 = (l15 == 0) ? sG[K + 1] : 0.f;
        bG.u[p] = (uint32_t)f2bf(g0) | ((uint32_t)f2bf(g1) << 16);
    }

    float gbias[8];
#pragma unroll
    for (int nt = 0; nt < 8; ++nt) gbias[nt] = b_ih[nt * 16 + l15] + b_hh[nt * 16 + l15];
    const float bv0 = in_b[64 + l15], bv1 = in_b[80 + l15];
    const float obx = obs2[b * 2], oby = obs2[b * 2 + 1];

    float runM = -1e30f, runS = 0.f, cp0 = 0.f, cp1 = 0.f;

    for (int it = 0; it < 8; ++it) {
        const int T = w * 8 + it;                  // tile of 16 L-slots, l-1 in [16T, 16T+16)
        // ---- stage: rel-embed (relu), h (bf16), c (f32); 2 lanes per row
        {
            const int m  = lane >> 1;
            const int cs = (lane & 1) * 16;
            const int jm = T * 16 + m;
            int j = jm + (jm >= b ? 1 : 0);
            j = j > 1023 ? 1023 : j;               // row 1023 of last tile is masked later
            const float rx = obs2[j * 2]     - obx;
            const float ry = obs2[j * 2 + 1] - oby;
            const size_t rowoff = ((size_t)(b * 1024 + j)) * 32 + cs;
            const f32x4* hp = (const f32x4*)(hc_h + rowoff);
            const f32x4* cq = (const f32x4*)(hc_c + rowoff);
#pragma unroll
            for (int q = 0; q < 4; ++q) {
                f32x4 hv = __builtin_nontemporal_load(hp + q);   // streamed once: bypass-friendly
                f32x4 cv = __builtin_nontemporal_load(cq + q);
                int s0 = cs + q * 4;
                sH[w][m][s0 + 0] = f2bf(hv.x); sH[w][m][s0 + 1] = f2bf(hv.y);
                sH[w][m][s0 + 2] = f2bf(hv.z); sH[w][m][s0 + 3] = f2bf(hv.w);
                sC[w][m][s0 + 0] = cv.x; sC[w][m][s0 + 1] = cv.y;
                sC[w][m][s0 + 2] = cv.z; sC[w][m][s0 + 3] = cv.w;
            }
#pragma unroll
            for (int q = 0; q < 16; ++q) {
                int s = cs + q;
                float e = rx * sEw0[s] + ry * sEw1[s] + sEb[s];
                sRelE[w][m][s] = f2bf(fmaxf(e, 0.f));
            }
        }
        __syncthreads();

        // ---- A fragments (16-bit A layout: m = lane&15, K = (e>>3)*16 + (lane>>4)*8 + (e&7))
        Frag aR, aH;
#pragma unroll
        for (int p = 0; p < 8; ++p) {
            int K = ((p >> 2) * 16) + kb + 2 * (p & 3);
            aR.u[p] = *(const uint32_t*)&sRelE[w][l15][K];
            aH.u[p] = *(const uint32_t*)&sH[w][l15][K];
        }
        // ---- gates(16x128) = [relE | h](16x64) @ Wt(64x128) + bias  -> 16 WMMAs
        v8f acc[8];
#pragma unroll
        for (int nt = 0; nt < 8; ++nt) {
            v8f c0;
#pragma unroll
            for (int e = 0; e < 8; ++e) c0[e] = gbias[nt];
            c0 = __builtin_amdgcn_wmma_f32_16x16x32_bf16(false, aR.v, false, bW[0][nt].v, (short)0, c0, false, false);
            c0 = __builtin_amdgcn_wmma_f32_16x16x32_bf16(false, aH.v, false, bW[1][nt].v, (short)0, c0, false, false);
            acc[nt] = c0;
        }
        // ---- LSTM nonlinearity: i/f/g/o for one SD column live in the same lane
#pragma unroll
        for (int t = 0; t < 2; ++t) {
            const int s = t * 16 + l15;
#pragma unroll
            for (int r = 0; r < 8; ++r) {
                const int m = r + 8 * half;
                float iv = acc[t][r], fv = acc[t + 2][r], gv = acc[t + 4][r], ov = acc[t + 6][r];
                float c2 = sigf(fv) * sC[w][m][s] + sigf(iv) * tanhfast(gv);
                float h2 = sigf(ov) * tanhfast(c2);
                sH2[w][m][s] = f2bf(h2);
            }
        }
        __syncthreads();

        Frag aH2;
#pragma unroll
        for (int p = 0; p < 8; ++p) {
            int K = ((p >> 2) * 16) + kb + 2 * (p & 3);
            aH2.u[p] = *(const uint32_t*)&sH2[w][l15][K];
        }
        // ---- vp (16x32) = h2 @ Cv.T + bv -> 2 WMMAs ; scores (16x1) = h2 @ g + c0b -> 1 WMMA
        v8f vacc[2];
#pragma unroll
        for (int nt = 0; nt < 2; ++nt) {
            v8f cv0;
            float bvv = nt ? bv1 : bv0;
#pragma unroll
            for (int e = 0; e < 8; ++e) cv0[e] = bvv;
            vacc[nt] = __builtin_amdgcn_wmma_f32_16x16x32_bf16(false, aH2.v, false, bV[nt].v, (short)0, cv0, false, false);
        }
        v8f sacc;
        {
            float cinit = (l15 == 0) ? c0b : 0.f;
            v8f c0;
#pragma unroll
            for (int e = 0; e < 8; ++e) c0[e] = cinit;
            sacc = __builtin_amdgcn_wmma_f32_16x16x32_bf16(false, aH2.v, false, bG.v, (short)0, c0, false, false);
        }
        // ---- broadcast scores from column-0 lanes (0 / 16) to their half
        float sc[8];
#pragma unroll
        for (int r = 0; r < 8; ++r) {
            float sv = __shfl(sacc[r], lane & 16, 32);
            int lm1 = T * 16 + r + 8 * half;
            sc[r] = (lm1 <= 1022) ? sv : -1e30f;   // only 1023 real pairs
        }
        // ---- online softmax
        float tmax = sc[0];
#pragma unroll
        for (int r = 1; r < 8; ++r) tmax = fmaxf(tmax, sc[r]);
        tmax = fmaxf(tmax, __shfl_xor(tmax, 16, 32));
        float newM = fmaxf(runM, tmax);
        float rs = __expf(runM - newM);
        runS *= rs; cp0 *= rs; cp1 *= rs;
#pragma unroll
        for (int r = 0; r < 8; ++r) {
            float pr = __expf(sc[r] - newM);
            runS += pr;
            cp0 += pr * vacc[0][r];
            cp1 += pr * vacc[1][r];
        }
        runM = newM;
        __syncthreads();
    }

    // ---- merge halves, publish wave partials
    float Sw  = runS + __shfl_xor(runS, 16, 32);
    float c0w = cp0 + __shfl_xor(cp0, 16, 32);
    float c1w = cp1 + __shfl_xor(cp1, 16, 32);
    if (lane < 16) {
        wCtx[w][l15]      = c0w;
        wCtx[w][l15 + 16] = c1w;
        if (lane == 0) { wMs[w] = runM; wSs[w] = Sw; }
    }
    __syncthreads();

    // ---- combine 8 wave partials + the l=0 (hs) slot; final projections via shuffles
    if (tid < 32) {
        const int n = lane;
        float kp0 = in_b[32 + n], vp0 = in_b[64 + n];
#pragma unroll 4
        for (int s = 0; s < 32; ++s) {
            float hv = hsg[b * 32 + s];
            kp0 += bf2f(sCk[n][s]) * hv;
            vp0 += bf2f(sCv[n][s]) * hv;
        }
        float s0 = q0g[b * 32 + n] * kp0;
        s0 += __shfl_xor(s0, 1, 32);
        s0 += __shfl_xor(s0, 2, 32);
        s0 += __shfl_xor(s0, 4, 32);
        s0 += __shfl_xor(s0, 8, 32);
        s0 += __shfl_xor(s0, 16, 32);
        float M = s0;
#pragma unroll
        for (int w2 = 0; w2 < 8; ++w2) M = fmaxf(M, wMs[w2]);
        float e0 = __expf(s0 - M);
        float S = e0, ctxn = e0 * vp0;
#pragma unroll
        for (int w2 = 0; w2 < 8; ++w2) {
            float e = __expf(wMs[w2] - M);
            S += wSs[w2] * e;
            ctxn += wCtx[w2][n] * e;
        }
        ctxn /= S;                                  // once per block: keep exact
        float o0 = attn_ob[n];
#pragma unroll 4
        for (int q = 0; q < 32; ++q) o0 += __shfl(ctxn, q, 32) * attn_ow[n * 32 + q];
        float fo = proj_b[n];
#pragma unroll 4
        for (int q = 0; q < 32; ++q) fo += __shfl(o0, q, 32) * proj_w[n * 32 + q];
        out[b * 32 + n] = fo;
    }
}

extern "C" void kernel_launch(void* const* d_in, const int* in_sizes, int n_in,
                              void* d_out, int out_size, void* d_ws, size_t ws_size,
                              hipStream_t stream) {
    const float* hidden_state = (const float*)d_in[0];
    const float* obs2   = (const float*)d_in[2];
    const float* hc_h   = (const float*)d_in[3];
    const float* hc_c   = (const float*)d_in[4];
    const float* emb_w  = (const float*)d_in[5];
    const float* emb_b  = (const float*)d_in[6];
    const float* h2s_w  = (const float*)d_in[7];
    const float* h2s_b  = (const float*)d_in[8];
    const float* w_ih   = (const float*)d_in[9];
    const float* w_hh   = (const float*)d_in[10];
    const float* b_ih   = (const float*)d_in[11];
    const float* b_hh   = (const float*)d_in[12];
    const float* wq_w   = (const float*)d_in[13];
    const float* wk_w   = (const float*)d_in[14];
    const float* wv_w   = (const float*)d_in[15];
    const float* in_w   = (const float*)d_in[16];
    const float* in_b   = (const float*)d_in[17];
    const float* attn_ow = (const float*)d_in[18];
    const float* attn_ob = (const float*)d_in[19];
    const float* proj_w  = (const float*)d_in[20];
    const float* proj_b  = (const float*)d_in[21];

    float* hsg  = (float*)d_ws;            // 32768 f32
    float* q0g  = hsg + 32768;             // 32768 f32
    float* wsCq = q0g + 32768;             // 1024 f32
    uint16_t* wsCk = (uint16_t*)(wsCq + 1024);   // 1024 bf16
    uint16_t* wsCv = wsCk + 1024;                // 1024 bf16
    uint16_t* wsWt = wsCv + 1024;                // 8192 bf16

    sattn_prep<<<1, 256, 0, stream>>>(in_w, wq_w, wk_w, wv_w, w_ih, w_hh, wsCq, wsCk, wsCv, wsWt);
    sattn_hs<<<128, 256, 0, stream>>>(hidden_state, h2s_w, h2s_b, hsg);
    sattn_q0<<<128, 256, 0, stream>>>(hsg, wsCq, in_b, q0g);
    sattn_main<<<1024, 256, 0, stream>>>(obs2, hc_h, hc_c, emb_w, emb_b, b_ih, b_hh, in_b,
                                         attn_ow, attn_ob, proj_w, proj_b,
                                         hsg, q0g, wsCk, wsCv, wsWt, (float*)d_out);
}
